// MultilayerGRU_17832704213004
// MI455X (gfx1250) — matrix operand order repaired and checked by reading, not compiled, not run on hardware
//
#include <hip/hip_runtime.h>
#include <cmath>

// ---------------------------------------------------------------------------
// CDNA5 (gfx1250) wave32 WMMA types
// ---------------------------------------------------------------------------
typedef __attribute__((ext_vector_type(16))) __bf16          v16bf;
typedef __attribute__((ext_vector_type(8)))  float           v8f;
typedef __attribute__((ext_vector_type(16))) unsigned short  v16us;
typedef __attribute__((ext_vector_type(8)))  unsigned short  v8us;
typedef __attribute__((ext_vector_type(4)))  unsigned int    v4u;
typedef __attribute__((ext_vector_type(8)))  int             v8i;
typedef __attribute__((ext_vector_type(4)))  int             v4i;

#define D_DIM 1024
#define S_LEN 2048
#define B_SZ  8
#define NBLK_GRU 64          // one block per 16-column tile: 1024/16

// LDS weight cache: 9 slices of 16 rows x 1024 bf16, rows padded to 1032 to
// stagger LDS banks (stride 2064B -> 4-bank shift per row for b128 reads).
#define LDSW     1032
#define W_SLICE  (16 * LDSW)                 // ushorts per slice
#define N_SLICES 9
#define SM_W_US  (N_SLICES * W_SLICE)        // 148,608 us = 297,216 B
#define SM_NFLT  1152                        // 768 acc + 128 z + 128 z1 + 128 xg
#define GRU_SMEM_BYTES (SM_W_US * 2 + SM_NFLT * 4)   // 301,824 B < 320 KB

static __device__ __forceinline__ unsigned short f32_to_bf16_rne(float f) {
    unsigned u = __builtin_bit_cast(unsigned, f);
    unsigned rnd = 0x7FFFu + ((u >> 16) & 1u);
    return (unsigned short)((u + rnd) >> 16);
}

static __device__ __forceinline__ float sigmoidf_fast(float x) {
    return 1.0f / (1.0f + __expf(-x));
}

// A fragment: 16x32 bf16: lane m=lane&15; lanes 0-15 hold
// K=[0..15]; lanes 16-31 hold K=[16..31].
static __device__ __forceinline__ v16bf load_a_frag(const unsigned short* A, int lda,
                                                    int kblock, int rowLimit) {
    const int lane = threadIdx.x & 31;
    const int row  = lane & 15;
    const int kb   = (lane < 16) ? 0 : 16;
    v16us r = {0,0,0,0,0,0,0,0,0,0,0,0,0,0,0,0};
    if (row < rowLimit) {
        const v8us* p = (const v8us*)(A + (size_t)row * lda + kblock + kb);
        v8us lo = p[0];
        v8us hi = p[1];
        r = __builtin_shufflevector(lo, hi, 0,1,2,3,4,5,6,7,8,9,10,11,12,13,14,15);
    }
    return __builtin_bit_cast(v16bf, r);
}

// B fragment from global: B[k][n] = W[col0+n][k], W row-major [N][K].
static __device__ __forceinline__ v16bf load_b_frag(const unsigned short* W, int ldw,
                                                    int col0, int kblock) {
    const int lane = threadIdx.x & 31;
    const int n    = lane & 15;
    const int kb   = (lane < 16) ? 0 : 16;
    const v8us* p  = (const v8us*)(W + (size_t)(col0 + n) * ldw + kblock + kb);
    v8us lo = p[0];
    v8us hi = p[1];
    v16us r = __builtin_shufflevector(lo, hi, 0,1,2,3,4,5,6,7,8,9,10,11,12,13,14,15);
    return __builtin_bit_cast(v16bf, r);
}

// B fragment from the LDS weight cache (local 16-row slice, padded stride).
static __device__ __forceinline__ v16bf load_b_frag_lds(const unsigned short* Ws, int kblock) {
    const int lane = threadIdx.x & 31;
    const int n    = lane & 15;
    const int kb   = (lane < 16) ? 0 : 16;
    const v8us* p  = (const v8us*)(Ws + (size_t)n * LDSW + kblock + kb);
    v8us lo = p[0];
    v8us hi = p[1];
    v16us r = __builtin_shufflevector(lo, hi, 0,1,2,3,4,5,6,7,8,9,10,11,12,13,14,15);
    return __builtin_bit_cast(v16bf, r);
}

// Partial-K accumulation: A from global (bf16, lda=1024), B from LDS cache.
static __device__ __forceinline__ v8f gemm_chunk(v8f acc, const unsigned short* Aglob,
                                                 const unsigned short* Wlds,
                                                 int k0, int klen) {
    for (int k = k0; k < k0 + klen; k += 32) {
        v16bf a = load_a_frag(Aglob, D_DIM, k, B_SZ);
        v16bf b = load_b_frag_lds(Wlds, k);
        acc = __builtin_amdgcn_wmma_f32_16x16x32_bf16(false, a, false, b,
                                                      (short)0, acc, false, false);
    }
    return acc;
}

// Reduce a wave's partial accumulator into the LDS 16x16 f32 tile (ds_add_f32).
static __device__ __forceinline__ void reduce_to_lds(float* accS, v8f acc) {
    const int lane = threadIdx.x & 31;
    const int n    = lane & 15;
    const int mh   = (lane < 16) ? 0 : 8;
#pragma unroll
    for (int v = 0; v < 8; ++v) {
        atomicAdd(&accS[(mh + v) * 16 + n], acc[v]);
    }
}

// ---------------------------------------------------------------------------
// f32 -> bf16 conversion
// ---------------------------------------------------------------------------
__global__ void k_cvt_bf16(const float* __restrict__ src, unsigned short* __restrict__ dst, int n) {
    int i = blockIdx.x * blockDim.x + threadIdx.x;
    if (i < n) dst[i] = f32_to_bf16_rne(src[i]);
}

// ---------------------------------------------------------------------------
// Big GEMM: C[M,N] = A[M,K](bf16) @ W[N,K](bf16)^T (+bias).
// grid = (M/16, N/64), block = 128 (4 waves, one 16-col tile each).
// ---------------------------------------------------------------------------
__global__ void k_gemm_bf16(const unsigned short* __restrict__ A, int lda,
                            const unsigned short* __restrict__ W, int ldw,
                            float* __restrict__ Cf, unsigned short* __restrict__ Cbf,
                            const float* __restrict__ bias,
                            int M, int N, int K) {
    const int wave  = threadIdx.x >> 5;
    const int lane  = threadIdx.x & 31;
    const int mtile = blockIdx.x;
    const int col0  = (blockIdx.y * 4 + wave) * 16;
    const unsigned short* Arow = A + (size_t)mtile * 16 * lda;

    v8f acc = {0.f,0.f,0.f,0.f,0.f,0.f,0.f,0.f};
    for (int k = 0; k < K; k += 32) {
        v16bf a = load_a_frag(Arow, lda, k, 16);
        v16bf b = load_b_frag(W, ldw, col0, k);
        acc = __builtin_amdgcn_wmma_f32_16x16x32_bf16(false, a, false, b,
                                                      (short)0, acc, false, false);
    }

    const int n   = lane & 15;
    const int mh  = (lane < 16) ? 0 : 8;
    const int col = col0 + n;
    const float bv = bias ? bias[col] : 0.0f;
#pragma unroll
    for (int v = 0; v < 8; ++v) {
        int row = mtile * 16 + mh + v;
        float val = acc[v] + bv;
        size_t idx = (size_t)row * N + col;
        if (Cf)  Cf[idx]  = val;
        if (Cbf) Cbf[idx] = f32_to_bf16_rne(val);
    }
}

// ---------------------------------------------------------------------------
// State init / final-hidden writeback
// ---------------------------------------------------------------------------
__global__ void k_init_state(const float* __restrict__ hs,
                             float* __restrict__ h0f, float* __restrict__ h1f,
                             unsigned short* __restrict__ h0bf, unsigned short* __restrict__ h1bf) {
    int i = blockIdx.x * blockDim.x + threadIdx.x;
    if (i < B_SZ * D_DIM) {
        int b = i >> 10, d = i & 1023;
        float v0 = hs[(size_t)(b * 2 + 0) * D_DIM + d];
        float v1 = hs[(size_t)(b * 2 + 1) * D_DIM + d];
        h0f[i] = v0;  h1f[i] = v1;
        h0bf[i] = f32_to_bf16_rne(v0);
        h1bf[i] = f32_to_bf16_rne(v1);
    }
}

__global__ void k_write_hidden(const float* __restrict__ h0f, const float* __restrict__ h1f,
                               float* __restrict__ out) {
    int i = blockIdx.x * blockDim.x + threadIdx.x;
    if (i < B_SZ * D_DIM) {
        int b = i >> 10, d = i & 1023;
        out[(size_t)(b * 2 + 0) * D_DIM + d] = h0f[i];
        out[(size_t)(b * 2 + 1) * D_DIM + d] = h1f[i];
    }
}

// ---------------------------------------------------------------------------
// Grid-wide barrier across 64 blocks, s_sleep backoff.
// ---------------------------------------------------------------------------
static __device__ __forceinline__ void grid_barrier(unsigned* bar) {
    __syncthreads();
    __threadfence();
    if (threadIdx.x == 0) {
        unsigned gen  = atomicAdd(&bar[1], 0u);
        unsigned prev = atomicAdd(&bar[0], 1u);
        if (prev == (unsigned)(NBLK_GRU - 1)) {
            atomicExch(&bar[0], 0u);
            __threadfence();
            atomicAdd(&bar[1], 1u);
        } else {
            while (atomicAdd(&bar[1], 0u) == gen) {
                __builtin_amdgcn_s_sleep(1);
            }
        }
    }
    __syncthreads();
}

// ---------------------------------------------------------------------------
// Persistent GRU recurrence.
// 64 blocks x 256 threads (8 waves). Block owns a 16-column tile; its 9 weight
// slices (290 KB bf16) live in LDS for all 2048 steps, staged once by the TDM.
// Each wave covers a 128-wide K chunk of every product; partials reduced with
// ds_add_f32 into an LDS f32 tile, 128 threads do the gate math.
// ---------------------------------------------------------------------------
__global__ void k_gru_persistent(
    const float* __restrict__ xz0, const float* __restrict__ xr0, const float* __restrict__ xg0,
    const unsigned short* __restrict__ whz0, const unsigned short* __restrict__ whr0,
    const unsigned short* __restrict__ whg0,
    const unsigned short* __restrict__ wxz1, const unsigned short* __restrict__ wxr1,
    const unsigned short* __restrict__ wxg1,
    const unsigned short* __restrict__ whz1, const unsigned short* __restrict__ whr1,
    const unsigned short* __restrict__ whg1,
    const float* __restrict__ Bz, const float* __restrict__ Br, const float* __restrict__ Bg,
    float* __restrict__ h0f, float* __restrict__ h1f,
    unsigned short* __restrict__ h0bf, unsigned short* __restrict__ h1bf,
    unsigned short* __restrict__ rhbf,
    unsigned short* __restrict__ H1bf,
    unsigned* __restrict__ bar) {

    extern __shared__ unsigned short smem[];          // dynamic LDS, base offset 0
    float* fbase = (float*)(smem + SM_W_US);
    float* accS  = fbase;            // 3 x 16 x 16 f32 accumulation tiles
    float* zS    = fbase + 768;      // 128: layer0 z (phase A -> B)
    float* z1S   = fbase + 896;      // 128: layer1 z (phase C -> D)
    float* xgS   = fbase + 1024;     // 128: layer1 x-side g (phase C -> D)

    const int tid  = threadIdx.x;
    const int wv   = tid >> 5;
    const int col0 = blockIdx.x * 16;
    const int kch  = wv * 128;                        // this wave's K chunk

    const unsigned short* wmats[N_SLICES] = {
        whz0, whr0, whg0, wxz1, whz1, wxr1, whr1, wxg1, whg1 };

    // ---- Stage the block's 9 weight slices into LDS ----
#if __has_builtin(__builtin_amdgcn_tensor_load_to_lds)
    if (wv == 0) {
        for (int s = 0; s < N_SLICES; ++s) {
            unsigned long long ga =
                (unsigned long long)(uintptr_t)(wmats[s] + (size_t)col0 * D_DIM);
            unsigned ldsb = (unsigned)(s * W_SLICE * 2);      // bytes, dyn-LDS base 0
            // D# group0: count=1, lds_addr, global_addr, type=2 (ISA 8.3)
            v4u g0 = { 1u, ldsb, (unsigned)(ga & 0xFFFFFFFFu),
                       (unsigned)((ga >> 32) & 0x01FFFFFFu) | (2u << 30) };
            // D# group1 (ISA 8.4): data_size=1(2B), iterate_enable=1,
            // tensor_dim0=1024[79:48], tensor_dim1=16[111:80],
            // tile_dim0=1024[127:112], tile_dim1=1[143:128],
            // tensor_dim0_stride=1024[207:160]
            v8i g1 = { (int)((1u << 16) | (1u << 19)),
                       (int)(((unsigned)D_DIM & 0xFFFFu) << 16),
                       (int)(16u << 16),
                       (int)((unsigned)D_DIM << 16),
                       (int)1u,
                       (int)(unsigned)D_DIM,
                       0, 0 };
            // D# group2 (iterate mode, ISA 8.5): tensor_dim2=1,
            // lds_addr_increment=1032, global_addr_increment=1024,
            // iterate_count=15 (16 rows)
            v4i g2 = { 1, (int)LDSW, (int)D_DIM, (int)(15u << 16) };
            v4i g3 = { 0, 0, 0, 0 };
            v8i gx = { 0, 0, 0, 0, 0, 0, 0, 0 };   // 6-arg form: extra group (zeros)
            __builtin_amdgcn_tensor_load_to_lds(g0, g1, g2, g3, gx, 0);
        }
        __builtin_amdgcn_s_wait_tensorcnt(0);
    }
#else
    for (int i = tid; i < N_SLICES * 16 * D_DIM; i += 256) {
        int s  = i / (16 * D_DIM);
        int r  = (i / D_DIM) & 15;
        int kk = i & (D_DIM - 1);
        smem[s * W_SLICE + r * LDSW + kk] = wmats[s][(size_t)(col0 + r) * D_DIM + kk];
    }
#endif
    __syncthreads();

    const v8f zero = {0.f,0.f,0.f,0.f,0.f,0.f,0.f,0.f};

    for (int t = 0; t < S_LEN; ++t) {
        // ---- Phase A: layer0 z,r ----
        for (int i = tid; i < 768; i += 256) accS[i] = 0.f;
        __syncthreads();
        {
            v8f az = gemm_chunk(zero, h0bf, smem + 0 * W_SLICE, kch, 128);
            v8f ar = gemm_chunk(zero, h0bf, smem + 1 * W_SLICE, kch, 128);
            reduce_to_lds(accS,       az);
            reduce_to_lds(accS + 256, ar);
        }
        __syncthreads();
        if (tid < 128) {
            int m = tid >> 4, n = tid & 15, col = col0 + n;
            size_t prow = ((size_t)m * S_LEN + t) * D_DIM + col;
            float z = sigmoidf_fast(accS[(m)*16 + n]       + xz0[prow] + Bz[col]);
            float r = sigmoidf_fast(accS[256 + (m)*16 + n] + xr0[prow] + Br[col]);
            float h = h0f[m * D_DIM + col];
            zS[tid] = z;
            rhbf[m * D_DIM + col] = f32_to_bf16_rne(r * h);
        }
        grid_barrier(bar);

        // ---- Phase B: layer0 g + hidden update ----
        for (int i = tid; i < 256; i += 256) accS[i] = 0.f;
        __syncthreads();
        reduce_to_lds(accS, gemm_chunk(zero, rhbf, smem + 2 * W_SLICE, kch, 128));
        __syncthreads();
        if (tid < 128) {
            int m = tid >> 4, n = tid & 15, col = col0 + n;
            size_t prow = ((size_t)m * S_LEN + t) * D_DIM + col;
            float g  = tanhf(accS[m*16 + n] + xg0[prow] + Bg[col]);
            float h  = h0f[m * D_DIM + col];
            float hn = (1.0f - zS[tid]) * g + zS[tid] * h;
            h0f[m * D_DIM + col]  = hn;
            h0bf[m * D_DIM + col] = f32_to_bf16_rne(hn);
        }
        grid_barrier(bar);

        // ---- Phase C: layer1 z,r and x-side g (input = fresh h0) ----
        for (int i = tid; i < 768; i += 256) accS[i] = 0.f;
        __syncthreads();
        {
            v8f az1 = gemm_chunk(zero, h0bf, smem + 3 * W_SLICE, kch, 128);
            az1     = gemm_chunk(az1,  h1bf, smem + 4 * W_SLICE, kch, 128);
            v8f ar1 = gemm_chunk(zero, h0bf, smem + 5 * W_SLICE, kch, 128);
            ar1     = gemm_chunk(ar1,  h1bf, smem + 6 * W_SLICE, kch, 128);
            v8f axg = gemm_chunk(zero, h0bf, smem + 7 * W_SLICE, kch, 128);
            reduce_to_lds(accS,       az1);
            reduce_to_lds(accS + 256, ar1);
            reduce_to_lds(accS + 512, axg);
        }
        __syncthreads();
        if (tid < 128) {
            int m = tid >> 4, n = tid & 15, col = col0 + n;
            float z = sigmoidf_fast(accS[m*16 + n]       + Bz[D_DIM + col]);
            float r = sigmoidf_fast(accS[256 + m*16 + n] + Br[D_DIM + col]);
            float h = h1f[m * D_DIM + col];
            z1S[tid] = z;
            xgS[tid] = accS[512 + m*16 + n];
            rhbf[m * D_DIM + col] = f32_to_bf16_rne(r * h);
        }
        grid_barrier(bar);

        // ---- Phase D: layer1 g + update + stash h1(t) ----
        for (int i = tid; i < 256; i += 256) accS[i] = 0.f;
        __syncthreads();
        reduce_to_lds(accS, gemm_chunk(zero, rhbf, smem + 8 * W_SLICE, kch, 128));
        __syncthreads();
        if (tid < 128) {
            int m = tid >> 4, n = tid & 15, col = col0 + n;
            float g  = tanhf(accS[m*16 + n] + xgS[tid] + Bg[D_DIM + col]);
            float h  = h1f[m * D_DIM + col];
            float hn = (1.0f - z1S[tid]) * g + z1S[tid] * h;
            h1f[m * D_DIM + col] = hn;
            unsigned short hb = f32_to_bf16_rne(hn);
            h1bf[m * D_DIM + col] = hb;
            H1bf[((size_t)m * S_LEN + t) * D_DIM + col] = hb;
        }
        grid_barrier(bar);
    }
}

// ---------------------------------------------------------------------------
// Host launcher
// ---------------------------------------------------------------------------
extern "C" void kernel_launch(void* const* d_in, const int* in_sizes, int n_in,
                              void* d_out, int out_size, void* d_ws, size_t ws_size,
                              hipStream_t stream) {
    (void)in_sizes; (void)n_in; (void)out_size; (void)ws_size;
    const float* X   = (const float*)d_in[0];
    const float* HS  = (const float*)d_in[1];
    const float* Wxz = (const float*)d_in[2];
    const float* Whz = (const float*)d_in[3];
    const float* Wxr = (const float*)d_in[4];
    const float* Whr = (const float*)d_in[5];
    const float* Wxg = (const float*)d_in[6];
    const float* Whg = (const float*)d_in[7];
    const float* Bz  = (const float*)d_in[8];
    const float* Br  = (const float*)d_in[9];
    const float* Bg  = (const float*)d_in[10];
    const float* Why = (const float*)d_in[11];
    const float* By  = (const float*)d_in[12];

    const size_t MROWS = (size_t)B_SZ * S_LEN;       // 16384
    const size_t MAT   = (size_t)D_DIM * D_DIM;      // 1M elems

    char* w = (char*)d_ws;
    auto carve = [&](size_t bytes) -> char* {
        char* p = w;
        w += (bytes + 255) & ~(size_t)255;
        return p;
    };

    unsigned*       bar   = (unsigned*)      carve(256);
    unsigned short* Xbf   = (unsigned short*)carve(MROWS * D_DIM * 2);
    unsigned short* wxzbf = (unsigned short*)carve(2 * MAT * 2);
    unsigned short* whzbf = (unsigned short*)carve(2 * MAT * 2);
    unsigned short* wxrbf = (unsigned short*)carve(2 * MAT * 2);
    unsigned short* whrbf = (unsigned short*)carve(2 * MAT * 2);
    unsigned short* wxgbf = (unsigned short*)carve(2 * MAT * 2);
    unsigned short* whgbf = (unsigned short*)carve(2 * MAT * 2);
    unsigned short* whybf = (unsigned short*)carve(MAT * 2);
    float*          xz0   = (float*)         carve(MROWS * D_DIM * 4);
    float*          xr0   = (float*)         carve(MROWS * D_DIM * 4);
    float*          xg0   = (float*)         carve(MROWS * D_DIM * 4);
    unsigned short* H1bf  = (unsigned short*)carve(MROWS * D_DIM * 2);
    float*          h0f   = (float*)         carve(B_SZ * D_DIM * 4);
    float*          h1f   = (float*)         carve(B_SZ * D_DIM * 4);
    unsigned short* h0bf  = (unsigned short*)carve(B_SZ * D_DIM * 2);
    unsigned short* h1bf  = (unsigned short*)carve(B_SZ * D_DIM * 2);
    unsigned short* rhbf  = (unsigned short*)carve(B_SZ * D_DIM * 2);

    (void)hipMemsetAsync(bar, 0, 256, stream);

    auto cvt = [&](const float* s, unsigned short* d, size_t n) {
        k_cvt_bf16<<<dim3((unsigned)((n + 255) / 256)), dim3(256), 0, stream>>>(s, d, (int)n);
    };
    cvt(X,   Xbf,   MROWS * D_DIM);
    cvt(Wxz, wxzbf, 2 * MAT);
    cvt(Whz, whzbf, 2 * MAT);
    cvt(Wxr, wxrbf, 2 * MAT);
    cvt(Whr, whrbf, 2 * MAT);
    cvt(Wxg, wxgbf, 2 * MAT);
    cvt(Whg, whgbf, 2 * MAT);
    cvt(Why, whybf, MAT);

    k_init_state<<<dim3((B_SZ * D_DIM + 255) / 256), dim3(256), 0, stream>>>(
        HS, h0f, h1f, h0bf, h1bf);

    // Hoisted layer-0 x-side GEMMs: [16384,1024] @ [1024,1024]^T
    dim3 ggrid((unsigned)(MROWS / 16), (unsigned)(D_DIM / 64));
    k_gemm_bf16<<<ggrid, 128, 0, stream>>>(Xbf, D_DIM, wxzbf, D_DIM,
                                           xz0, nullptr, nullptr,
                                           (int)MROWS, D_DIM, D_DIM);
    k_gemm_bf16<<<ggrid, 128, 0, stream>>>(Xbf, D_DIM, wxrbf, D_DIM,
                                           xr0, nullptr, nullptr,
                                           (int)MROWS, D_DIM, D_DIM);
    k_gemm_bf16<<<ggrid, 128, 0, stream>>>(Xbf, D_DIM, wxgbf, D_DIM,
                                           xg0, nullptr, nullptr,
                                           (int)MROWS, D_DIM, D_DIM);

    // Persistent recurrence (layer-1 weights = second half of stacked arrays)
    k_gru_persistent<<<dim3(NBLK_GRU), dim3(256), GRU_SMEM_BYTES, stream>>>(
        xz0, xr0, xg0,
        whzbf, whrbf, whgbf,                        // layer-0 h-side
        wxzbf + MAT, wxrbf + MAT, wxgbf + MAT,      // layer-1 x-side
        whzbf + MAT, whrbf + MAT, whgbf + MAT,      // layer-1 h-side
        Bz, Br, Bg,
        h0f, h1f, h0bf, h1bf, rhbf, H1bf, bar);

    // Deferred output projection: Y = H1 @ Why^T + By -> d_out[0 .. B*S*D)
    k_gemm_bf16<<<ggrid, 128, 0, stream>>>(H1bf, D_DIM, whybf, D_DIM,
                                           (float*)d_out, nullptr, By,
                                           (int)MROWS, D_DIM, D_DIM);

    // final_hidden [B,L,D] appended after layer_output
    k_write_hidden<<<dim3((B_SZ * D_DIM + 255) / 256), dim3(256), 0, stream>>>(
        h0f, h1f, (float*)d_out + MROWS * D_DIM);
}
